// Transition_5841155522944
// MI455X (gfx1250) — matrix operand order repaired and checked by reading, not compiled, not run on hardware
//
#include <hip/hip_runtime.h>

// ---------------------------------------------------------------------------
// Binary-weight 3x3 conv (pad 1) + BN + binary_tanh for MI455X (gfx1250).
// Implicit GEMM:  M = Cout = 128, N = B*H*W = 204800, K = Cin*9 = 1152.
// bf16 WMMA (v_wmma_f32_16x16x32_bf16), fp32 accumulation.
//
// v2: weights sign-binarized ONCE into d_ws (bf16, [co][K] row-major) by a
//     pre-kernel; A-fragments then load straight from global (no LDS, no
//     per-chunk re-binarization). im2col B gather is software-pipelined:
//     next chunk's global loads overlap current chunk's WMMAs.
// ---------------------------------------------------------------------------

typedef __attribute__((ext_vector_type(16))) __bf16 v16bf;
typedef __attribute__((ext_vector_type(8)))  float  v8f;

#define CIN    128
#define COUT   128
#define HDIM   40
#define WDIM   40
#define HW     1600      // 40*40
#define KTOT   1152      // CIN*9
#define KC     64        // K-chunk per iteration (2 WMMA K-steps)
#define NKC    (KTOT/KC) // 18 chunks
#define NT     64        // pixels (N) per workgroup
#define NTHR   256       // 8 waves of 32
#define WBYTES ((size_t)COUT * KTOT * 2)   // 294912 B of binarized weights

// float -> bf16 bits, round-to-nearest-even
__device__ __forceinline__ unsigned short f2bf(float f) {
    unsigned u = __float_as_uint(f);
    unsigned r = u + 0x7FFFu + ((u >> 16) & 1u);
    return (unsigned short)(r >> 16);
}

// sign(w) as bf16 bits: +1 / -1 / 0
__device__ __forceinline__ unsigned short sbits(float w) {
    return (w > 0.f) ? (unsigned short)0x3F80u
                     : ((w < 0.f) ? (unsigned short)0xBF80u : (unsigned short)0u);
}

// ---- pre-kernel: binarize all weights to bf16 bits in scratch -------------
__global__ __launch_bounds__(256)
void binarize_weights(const float* __restrict__ wgt, unsigned short* __restrict__ ws) {
    const int i = blockIdx.x * 256 + threadIdx.x;
    if (i < COUT * KTOT) ws[i] = sbits(wgt[i]);
}

// im2col gather of this thread's 16 B-tile elements for K index kg = k0+kl
__device__ __forceinline__ void gather_b(const float* __restrict__ x, int b, int p0,
                                         int sub, unsigned kg, float* __restrict__ v) {
    const unsigned ci = kg / 9u;
    const unsigned rr = kg - ci * 9u;
    const int kh = (int)(rr / 3u) - 1;                 // -1..1
    const int kw = (int)(rr - (rr / 3u) * 3u) - 1;     // -1..1
    const float* xb = x + ((size_t)b * CIN + ci) * HW;
#pragma unroll
    for (int i = 0; i < 16; ++i) {
        const int n  = i * 4 + sub;                    // 0..63
        const int p  = p0 + n;
        const int h  = p / WDIM;
        const int w  = p - h * WDIM;
        const int ih = h + kh;
        const int iw = w + kw;
        float val = 0.f;
        if ((unsigned)ih < (unsigned)HDIM && (unsigned)iw < (unsigned)WDIM)
            val = xb[ih * WDIM + iw];
        v[i] = val;
    }
}

// Load one A fragment (16x32 bf16, wave32 layout) for this lane.
// off = element offset within the K row: chunks [off, off+8) and [off+16, off+24).
template <bool PRE>
__device__ __forceinline__ v16bf load_afrag(const unsigned short* __restrict__ wsrow,
                                            const float* __restrict__ wrow, int off) {
    union { uint4 u[2]; v16bf v; } ua;
    if (PRE) {
        ua.u[0] = *(const uint4*)(wsrow + off);
        ua.u[1] = *(const uint4*)(wsrow + off + 16);
    } else {
        const float4 a0 = *(const float4*)(wrow + off);
        const float4 a1 = *(const float4*)(wrow + off + 4);
        const float4 b0 = *(const float4*)(wrow + off + 16);
        const float4 b1 = *(const float4*)(wrow + off + 20);
        ua.u[0].x = (unsigned)sbits(a0.x) | ((unsigned)sbits(a0.y) << 16);
        ua.u[0].y = (unsigned)sbits(a0.z) | ((unsigned)sbits(a0.w) << 16);
        ua.u[0].z = (unsigned)sbits(a1.x) | ((unsigned)sbits(a1.y) << 16);
        ua.u[0].w = (unsigned)sbits(a1.z) | ((unsigned)sbits(a1.w) << 16);
        ua.u[1].x = (unsigned)sbits(b0.x) | ((unsigned)sbits(b0.y) << 16);
        ua.u[1].y = (unsigned)sbits(b0.z) | ((unsigned)sbits(b0.w) << 16);
        ua.u[1].z = (unsigned)sbits(b1.x) | ((unsigned)sbits(b1.y) << 16);
        ua.u[1].w = (unsigned)sbits(b1.z) | ((unsigned)sbits(b1.w) << 16);
    }
    return ua.v;
}

template <bool PRE>
__global__ __launch_bounds__(NTHR)
void binconv_bn_btanh(const float* __restrict__ x,
                      const float* __restrict__ wgt,
                      const unsigned short* __restrict__ ws,
                      const float* __restrict__ gamma,
                      const float* __restrict__ beta,
                      const float* __restrict__ rmean,
                      const float* __restrict__ rvar,
                      float* __restrict__ out)
{
    __shared__ __align__(16) unsigned short Bs[NT * KC];   // 8 KB im2col tile (bf16 bits)
    __shared__ float scaleS[COUT];
    __shared__ float shiftS[COUT];

    const int tid  = threadIdx.x;
    const int lane = tid & 31;
    const int wv   = tid >> 5;          // wave id 0..7 -> Cout rows [16wv, 16wv+16)
    const int b    = blockIdx.y;
    const int p0   = blockIdx.x * NT;

    if (tid < COUT) {
        const float inv = rsqrtf(rvar[tid] + 1e-5f);
        const float sc  = gamma[tid] * inv;
        scaleS[tid] = sc;
        shiftS[tid] = beta[tid] - rmean[tid] * sc;
    }

    v8f acc[4];
#pragma unroll
    for (int t = 0; t < 4; ++t)
        acc[t] = (v8f){0.f,0.f,0.f,0.f,0.f,0.f,0.f,0.f};

    // B-staging thread mapping: fixed K lane, 16 pixels per thread
    const int kl  = tid & 63;           // K offset within chunk
    const int sub = tid >> 6;           // 0..3

    // Fragment addressing (wave32 WMMA VGPR layouts, ISA 7.12.2)
    const int m_row = lane & 15;
    const int ka    = (lane < 16) ? 0 : 8;    // A K-base per half-wave
    const int kb    = (lane < 16) ? 0 : 8;    // B K-base per half-wave

    const int co_a = wv * 16 + m_row;                        // A row this lane feeds
    const unsigned short* wsrow = PRE ? (ws + (size_t)co_a * KTOT) : nullptr;
    const float*          wrow  = PRE ? nullptr : (wgt + (size_t)co_a * KTOT);

    // Pipeline: gather chunk 0 into registers
    float vbuf[16];
    gather_b(x, b, p0, sub, (unsigned)kl, vbuf);

    for (int c = 0; c < NKC; ++c) {
        const int k0 = c * KC;

        // commit staged gather to LDS (fp32 -> bf16 bits)
#pragma unroll
        for (int i = 0; i < 16; ++i)
            Bs[(i * 4 + sub) * KC + kl] = f2bf(vbuf[i]);
        __syncthreads();                 // Bs ready (also covers scaleS/shiftS)

        // prefetch next chunk's gathers; global latency overlaps the WMMAs below
        if (c + 1 < NKC)
            gather_b(x, b, p0, sub, (unsigned)(k0 + KC + kl), vbuf);

        // A fragments straight from global (bf16 rows, 16B-aligned chunks)
        v16bf afrag[2];
#pragma unroll
        for (int s = 0; s < 2; ++s)
            afrag[s] = load_afrag<PRE>(wsrow, wrow, k0 + s * 32 + ka);

        // 2 K-steps x 4 N-tiles of v_wmma_f32_16x16x32_bf16 per wave
#pragma unroll
        for (int s = 0; s < 2; ++s) {
#pragma unroll
            for (int t = 0; t < 4; ++t) {
                const uint4* pb = (const uint4*)(Bs + (t * 16 + m_row) * KC + s * 32 + kb);
                union { uint4 u[2]; v16bf v; } ub;
                ub.u[0] = pb[0];
                ub.u[1] = pb[2];
                acc[t] = __builtin_amdgcn_wmma_f32_16x16x32_bf16(
                    false, afrag[s], false, ub.v, (short)0, acc[t], false, false);
            }
        }
        __syncthreads();                 // Bs consumed before next commit
    }

    // epilogue: BN + binary_tanh, store +-1.0f in NCHW
    // C/D layout: VGPR r -> M = r + (lane<16 ? 0 : 8); N = lane & 15.
    const int madd = (lane >> 4) << 3;
#pragma unroll
    for (int t = 0; t < 4; ++t) {
        const int p = p0 + t * 16 + (lane & 15);
#pragma unroll
        for (int r = 0; r < 8; ++r) {
            const int co = wv * 16 + madd + r;
            const float y = acc[t][r] * scaleS[co] + shiftS[co];
            const float o = (y > 0.f) ? 1.f : ((y < 0.f) ? -1.f : 0.f);
            out[((size_t)b * COUT + co) * HW + p] = o;
        }
    }
}

extern "C" void kernel_launch(void* const* d_in, const int* in_sizes, int n_in,
                              void* d_out, int out_size, void* d_ws, size_t ws_size,
                              hipStream_t stream) {
    (void)in_sizes; (void)n_in; (void)out_size;
    const float* x     = (const float*)d_in[0];
    const float* wgt   = (const float*)d_in[1];
    const float* gamma = (const float*)d_in[2];
    const float* beta  = (const float*)d_in[3];
    const float* rmean = (const float*)d_in[4];
    const float* rvar  = (const float*)d_in[5];
    float* out = (float*)d_out;

    dim3 grid(HW / NT, 128);             // 25 pixel-tiles x 128 images
    dim3 block(NTHR);

    if (ws_size >= WBYTES) {
        unsigned short* ws = (unsigned short*)d_ws;
        binarize_weights<<<(COUT * KTOT + 255) / 256, 256, 0, stream>>>(wgt, ws);
        binconv_bn_btanh<true><<<grid, block, 0, stream>>>(
            x, wgt, ws, gamma, beta, rmean, rvar, out);
    } else {
        binconv_bn_btanh<false><<<grid, block, 0, stream>>>(
            x, wgt, nullptr, gamma, beta, rmean, rvar, out);
    }
}
